// MultiHeadAttention_14413910245736
// MI455X (gfx1250) — matrix operand order repaired and checked by reading, not compiled, not run on hardware
//
#include <hip/hip_runtime.h>

// ---------------------------------------------------------------------------
// MI455X (gfx1250) causal MHA forward.
// bf16 WMMA (v_wmma_f32_16x16x32_bf16) everywhere, fp32 accumulate.
// B=2, S=2048, D=1024, H=16, HD=64.
// ---------------------------------------------------------------------------

typedef __attribute__((ext_vector_type(16))) __bf16        bf16x16;
typedef __attribute__((ext_vector_type(8)))  float         f32x8;
typedef __attribute__((ext_vector_type(4)))  unsigned int  u32x4;

union Frag {
  u32x4   q[2];   // 32 bytes = 16 bf16
  bf16x16 v;
};

__device__ __forceinline__ unsigned short f2bf(float x) {
  union { float f; unsigned int u; } c; c.f = x;
  unsigned int r = c.u + 0x7FFFu + ((c.u >> 16) & 1u);   // round-to-nearest-even
  return (unsigned short)(r >> 16);
}

__device__ __forceinline__ f32x8 wmma_bf16(const Frag& a, const Frag& b, f32x8 c) {
  // (neg_a, A, neg_b, B, c_mod, C, reuse_a, reuse_b)
  return __builtin_amdgcn_wmma_f32_16x16x32_bf16(false, a.v, false, b.v,
                                                 (short)0, c, false, false);
}

// ---------------------------------------------------------------------------
// fp32 -> bf16 conversion
// ---------------------------------------------------------------------------
__global__ void cvt_f32_bf16(const float* __restrict__ in,
                             unsigned short* __restrict__ out, int n) {
  int i = blockIdx.x * blockDim.x + threadIdx.x;
  if (i < n) out[i] = f2bf(in[i]);
}

// ---------------------------------------------------------------------------
// C = A(bf16 [M][1024]) x W(bf16 [N=1024][K=1024], torch layout) ^T + bias
// mode 0/1 : store bf16 to [B,H,S,64]      (Q, K)
// mode 2   : store bf16 to [B,H,64,S]      (V transposed for PV B-fragments)
// mode 3   : store fp32 to [M][1024]       (final projection)
// Block: 256 threads (8 waves), 128x128 tile, BK=32.
// Wave (2x4 grid) owns 64x32 -> 8 wmma tiles.
// ---------------------------------------------------------------------------
__global__ __launch_bounds__(256) void gemm_bf16_wmma(
    const unsigned short* __restrict__ A,
    const unsigned short* __restrict__ W,
    const float* __restrict__ bias,
    void* __restrict__ outp,
    int mode)
{
  constexpr int K  = 1024;
  constexpr int ST = 40;   // LDS row stride in bf16 (padded: 80B/row)
  __shared__ __align__(16) unsigned short As[128 * ST];
  __shared__ __align__(16) unsigned short Bs[128 * ST];

  const int tid  = threadIdx.x;
  const int lane = tid & 31;
  const int wave = tid >> 5;
  const int l15  = lane & 15;
  const int hi   = lane >> 4;
  const int m0   = blockIdx.x * 128;
  const int n0   = blockIdx.y * 128;
  const int wm   = (wave >> 2) * 64;
  const int wn   = (wave & 3) * 32;

  // cooperative tile loads: 2 threads per row, 32B each
  const int ldRow  = tid >> 1;
  const int ldHalf = (tid & 1) * 16;
  const unsigned short* gA = A + (size_t)(m0 + ldRow) * K + ldHalf;
  const unsigned short* gW = W + (size_t)(n0 + ldRow) * K + ldHalf;
  unsigned short* sA = As + ldRow * ST + ldHalf;
  unsigned short* sB = Bs + ldRow * ST + ldHalf;

  f32x8 acc[4][2] = {};

  for (int kk = 0; kk < K; kk += 32) {
    __syncthreads();
    {
      const u32x4* pa = (const u32x4*)(gA + kk);
      const u32x4* pw = (const u32x4*)(gW + kk);
      u32x4 a0 = pa[0], a1 = pa[1];
      u32x4 w0 = pw[0], w1 = pw[1];
      ((u32x4*)sA)[0] = a0; ((u32x4*)sA)[1] = a1;
      ((u32x4*)sB)[0] = w0; ((u32x4*)sB)[1] = w1;
    }
    __syncthreads();
    if (kk + 32 < K) {
      // high temporal locality -> near-scope prefetch (pull next K-tile toward
      // the WGP cache; data already resides in the 192MB L2)
      __builtin_prefetch(gA + kk + 32, 0, 3);
      __builtin_prefetch(gW + kk + 32, 0, 3);
    }

    Frag af[4], bf_[2];
    #pragma unroll
    for (int mi = 0; mi < 4; ++mi) {
      // A frag: lane<16 -> K 0..15; lane>=16 -> K 16..31
      const u32x4* p = (const u32x4*)(As + (wm + mi * 16 + l15) * ST + 16 * hi);
      af[mi].q[0] = p[0];
      af[mi].q[1] = p[1];
    }
    #pragma unroll
    for (int ni = 0; ni < 2; ++ni) {
      // B frag: lane<16 -> K 0..15 (col=lane); lane>=16 -> K 16..31 (col=lane-16)
      const u32x4* p = (const u32x4*)(Bs + (wn + ni * 16 + l15) * ST + 16 * hi);
      bf_[ni].q[0] = p[0];
      bf_[ni].q[1] = p[1];
    }
    #pragma unroll
    for (int mi = 0; mi < 4; ++mi)
      #pragma unroll
      for (int ni = 0; ni < 2; ++ni)
        acc[mi][ni] = wmma_bf16(af[mi], bf_[ni], acc[mi][ni]);
  }

  // epilogue: C layout — lane<16: col=lane, rows r; lane>=16: col=lane-16, rows r+8
  #pragma unroll
  for (int mi = 0; mi < 4; ++mi) {
    #pragma unroll
    for (int ni = 0; ni < 2; ++ni) {
      const int   nG  = n0 + wn + ni * 16 + l15;
      const float bia = bias[nG];
      #pragma unroll
      for (int r = 0; r < 8; ++r) {
        const int   mG  = m0 + wm + mi * 16 + r + 8 * hi;
        const float val = acc[mi][ni][r] + bia;
        if (mode == 3) {
          ((float*)outp)[(size_t)mG * 1024 + nG] = val;
        } else {
          const int bb = mG >> 11, ss = mG & 2047;
          const int hh = nG >> 6,  hd = nG & 63;
          size_t idx;
          if (mode == 2) idx = ((size_t)((bb * 16 + hh) * 64 + hd)) * 2048 + ss;
          else           idx = ((size_t)((bb * 16 + hh) * 2048 + ss)) * 64 + hd;
          ((unsigned short*)outp)[idx] = f2bf(val);
        }
      }
    }
  }
}

// ---------------------------------------------------------------------------
// Flash attention (causal). Q,K: bf16 [BH][S][64]; Vt: bf16 [BH][64][S].
// Each wave: 16 queries x HD=64; key blocks of 32; online softmax with
// 16-lane shfl reductions; P remapped C->A layout via per-wave LDS.
// No workgroup barriers (waves have divergent causal trip counts).
// ---------------------------------------------------------------------------
__global__ __launch_bounds__(256) void attn_wmma(
    const unsigned short* __restrict__ Q,
    const unsigned short* __restrict__ Km,
    const unsigned short* __restrict__ Vt,
    unsigned short* __restrict__ O)
{
  constexpr int S   = 2048;
  constexpr int PST = 40;
  __shared__ __align__(16) unsigned short Pl[8 * 16 * PST];

  const int lane  = threadIdx.x & 31;
  const int wave  = threadIdx.x >> 5;
  const int l15   = lane & 15;
  const int hi    = lane >> 4;
  const int bh    = blockIdx.y;
  const int qBase = (blockIdx.x * 8 + wave) * 16;

  const unsigned short* q  = Q  + (size_t)bh * S * 64;
  const unsigned short* k  = Km + (size_t)bh * S * 64;
  const unsigned short* vt = Vt + (size_t)bh * 64 * S;
  unsigned short* pw = Pl + wave * 16 * PST;

  Frag qf[2];                                   // Q 16x64 as two 16x32 A-frags
  #pragma unroll
  for (int kc = 0; kc < 2; ++kc) {
    const u32x4* p = (const u32x4*)(q + (size_t)(qBase + l15) * 64 + 32 * kc + 16 * hi);
    qf[kc].q[0] = p[0];
    qf[kc].q[1] = p[1];
  }

  f32x8 o[4] = {};
  float mrow[8], lrow[8];
  #pragma unroll
  for (int r = 0; r < 8; ++r) { mrow[r] = -3.0e38f; lrow[r] = 0.0f; }

  const float invTau    = 0.125f;      // 1/sqrt(64)
  const float negMasked = -1.25e8f;    // (-1e9)/tau : mask BEFORE /tau, as in ref

  for (int kbase = 0; kbase < qBase + 16; kbase += 32) {
    // ---- S = Q K^T for two 16-key tiles
    f32x8 sc[2];
    #pragma unroll
    for (int j = 0; j < 2; ++j) {
      f32x8 a = {};
      const int n = kbase + 16 * j + l15;        // key column for this lane
      Frag kf;
      const u32x4* p0 = (const u32x4*)(k + (size_t)n * 64 + 0 + 16 * hi);
      kf.q[0] = p0[0]; kf.q[1] = p0[1];
      a = wmma_bf16(qf[0], kf, a);
      const u32x4* p1 = (const u32x4*)(k + (size_t)n * 64 + 32 + 16 * hi);
      kf.q[0] = p1[0]; kf.q[1] = p1[1];
      a = wmma_bf16(qf[1], kf, a);
      sc[j] = a;
    }

    // ---- causal mask + online softmax (row r lives in one 16-lane half)
    float scaleF[8], p0v[8], p1v[8];
    #pragma unroll
    for (int r = 0; r < 8; ++r) {
      const int qi = qBase + r + 8 * hi;
      float s0 = (kbase + l15      <= qi) ? sc[0][r] * invTau : negMasked;
      float s1 = (kbase + 16 + l15 <= qi) ? sc[1][r] * invTau : negMasked;
      float mx = fmaxf(s0, s1);
      #pragma unroll
      for (int d = 8; d >= 1; d >>= 1) mx = fmaxf(mx, __shfl_xor(mx, d, 32));
      const float mnew = fmaxf(mrow[r], mx);
      const float e0 = __expf(s0 - mnew);
      const float e1 = __expf(s1 - mnew);
      float sm = e0 + e1;
      #pragma unroll
      for (int d = 8; d >= 1; d >>= 1) sm += __shfl_xor(sm, d, 32);
      const float sf = __expf(mrow[r] - mnew);
      lrow[r]   = lrow[r] * sf + sm;
      mrow[r]   = mnew;
      scaleF[r] = sf;
      p0v[r] = e0; p1v[r] = e1;
    }

    #pragma unroll
    for (int nt = 0; nt < 4; ++nt)
      #pragma unroll
      for (int r = 0; r < 8; ++r)
        o[nt][r] *= scaleF[r];

    // ---- P: C-layout regs -> per-wave LDS -> A-fragment (16x32 bf16)
    #pragma unroll
    for (int r = 0; r < 8; ++r) {
      const int pr = r + 8 * hi;
      pw[pr * PST + l15]      = f2bf(p0v[r]);
      pw[pr * PST + 16 + l15] = f2bf(p1v[r]);
    }
    __builtin_amdgcn_fence(__ATOMIC_RELEASE, "wavefront");
    __builtin_amdgcn_wave_barrier();
    __builtin_amdgcn_fence(__ATOMIC_ACQUIRE, "wavefront");

    Frag pf;
    {
      const u32x4* p = (const u32x4*)(pw + l15 * PST + 16 * hi);
      pf.q[0] = p[0];
      pf.q[1] = p[1];
    }

    // ---- O += P x V  (Vt rows are key-contiguous: perfect B-fragments)
    #pragma unroll
    for (int nt = 0; nt < 4; ++nt) {
      const u32x4* p = (const u32x4*)(vt + (size_t)(16 * nt + l15) * S + kbase + 16 * hi);
      Frag vf;
      vf.q[0] = p[0]; vf.q[1] = p[1];
      o[nt] = wmma_bf16(pf, vf, o[nt]);
    }
  }

  // ---- normalize + store bf16 to [B][S][D] (merge heads)
  const int bb = bh >> 4, hh = bh & 15;
  #pragma unroll
  for (int nt = 0; nt < 4; ++nt) {
    #pragma unroll
    for (int r = 0; r < 8; ++r) {
      const int ss   = qBase + r + 8 * hi;
      const int feat = hh * 64 + 16 * nt + l15;
      O[(size_t)(bb * 2048 + ss) * 1024 + feat] = f2bf(o[nt][r] / lrow[r]);
    }
  }
}

// ---------------------------------------------------------------------------
extern "C" void kernel_launch(void* const* d_in, const int* in_sizes, int n_in,
                              void* d_out, int out_size, void* d_ws, size_t ws_size,
                              hipStream_t stream) {
  (void)in_sizes; (void)n_in; (void)out_size; (void)ws_size;

  const float* x  = (const float*)d_in[0];
  // d_in[1] = causal mask (implicit in kernel)
  const float* Wq = (const float*)d_in[2];
  const float* bq = (const float*)d_in[3];
  const float* Wk = (const float*)d_in[4];
  const float* bk = (const float*)d_in[5];
  const float* Wv = (const float*)d_in[6];
  const float* bv = (const float*)d_in[7];
  const float* Wo = (const float*)d_in[8];
  const float* bo = (const float*)d_in[9];

  const size_t NX = (size_t)4096 * 1024;   // B*S*D
  const size_t NW = (size_t)1024 * 1024;   // D*D

  unsigned short* xb  = (unsigned short*)d_ws;     // bf16 activations
  unsigned short* wqb = xb  + NX;
  unsigned short* wkb = wqb + NW;
  unsigned short* wvb = wkb + NW;
  unsigned short* wob = wvb + NW;
  unsigned short* qb  = wob + NW;                  // [B,H,S,64]
  unsigned short* kb  = qb  + NX;                  // [B,H,S,64]
  unsigned short* vtb = kb  + NX;                  // [B,H,64,S]
  unsigned short* ob  = vtb + NX;                  // [B,S,D]

  cvt_f32_bf16<<<(unsigned)((NX + 255) / 256), 256, 0, stream>>>(x,  xb,  (int)NX);
  cvt_f32_bf16<<<(unsigned)((NW + 255) / 256), 256, 0, stream>>>(Wq, wqb, (int)NW);
  cvt_f32_bf16<<<(unsigned)((NW + 255) / 256), 256, 0, stream>>>(Wk, wkb, (int)NW);
  cvt_f32_bf16<<<(unsigned)((NW + 255) / 256), 256, 0, stream>>>(Wv, wvb, (int)NW);
  cvt_f32_bf16<<<(unsigned)((NW + 255) / 256), 256, 0, stream>>>(Wo, wob, (int)NW);

  dim3 gg(32, 8);   // M/128 x N/128
  gemm_bf16_wmma<<<gg, 256, 0, stream>>>(xb, wqb, bq, qb,  0);
  gemm_bf16_wmma<<<gg, 256, 0, stream>>>(xb, wkb, bk, kb,  1);
  gemm_bf16_wmma<<<gg, 256, 0, stream>>>(xb, wvb, bv, vtb, 2);

  attn_wmma<<<dim3(16, 32), 256, 0, stream>>>(qb, kb, vtb, ob);

  gemm_bf16_wmma<<<gg, 256, 0, stream>>>(ob, wob, bo, d_out, 3);
}
